// CoAttentionFusion_43782896615810
// MI455X (gfx1250) — matrix operand order repaired and checked by reading, not compiled, not run on hardware
//
#include <hip/hip_runtime.h>
#include <hip/hip_bf16.h>
#include <stdint.h>

// ---------------------------------------------------------------------------
// CDNA5 (gfx1250) WMMA types
// ---------------------------------------------------------------------------
typedef __attribute__((ext_vector_type(16))) __bf16 v16bf;
typedef __attribute__((ext_vector_type(8)))  float  v8f;

union FragAB { v16bf v; uint4 q[2]; uint16_t e[16]; };
union FragC  { v8f   v; float f[8]; };
union Pack8  { uint4 q; uint16_t e[8]; };

#define WMMA_BF16(a, b, c) \
  __builtin_amdgcn_wmma_f32_16x16x32_bf16(false, (a), false, (b), (short)0, (c), false, false)

__device__ __forceinline__ uint16_t f2bf(float x) {
  __hip_bfloat16 h = __float2bfloat16(x);
  return __builtin_bit_cast(uint16_t, h);
}

// ---------------------------------------------------------------------------
// f32 -> bf16 elementwise convert
// ---------------------------------------------------------------------------
__global__ void k_f32_to_bf16(const float* __restrict__ src,
                              uint16_t* __restrict__ dst, int n) {
  int i = blockIdx.x * 256 + threadIdx.x;
  if (i < n) dst[i] = f2bf(src[i]);
}

// ---------------------------------------------------------------------------
// bf16 transpose: in[R][C] -> out[C][R]   (R,C multiples of 32)
// ---------------------------------------------------------------------------
__global__ __launch_bounds__(256) void k_transpose_bf16(
    const uint16_t* __restrict__ in, uint16_t* __restrict__ out, int R, int C) {
  __shared__ uint16_t tile[32][33];
  const int c0 = blockIdx.x * 32, r0 = blockIdx.y * 32;
  const int tx = threadIdx.x & 31, ty = threadIdx.x >> 5;   // ty 0..7
  #pragma unroll
  for (int i = 0; i < 32; i += 8)
    tile[ty + i][tx] = in[(size_t)(r0 + ty + i) * C + c0 + tx];
  __syncthreads();
  #pragma unroll
  for (int i = 0; i < 32; i += 8)
    out[(size_t)(c0 + ty + i) * R + r0 + tx] = tile[tx][ty + i];
}

// ---------------------------------------------------------------------------
// bf16 GEMM:  C[M,N] = A[M,K] @ B[K,N] + bias, f32 accumulate via WMMA.
// Block tile 128x128, BK=32, 256 threads (8 wave32s), each wave 64x32.
// Double-buffered LDS: global loads for tile k+1 overlap the WMMAs of tile k.
// ---------------------------------------------------------------------------
#define BM 128
#define BN 128
#define BK 32
#define LSTR 40   // bf16 elems per LDS row: 80 bytes (16B aligned, conflict-free)

template <int OUT_BF16>
__global__ __launch_bounds__(256) void k_gemm_bf16(
    const uint16_t* __restrict__ A,    // M x K row-major bf16
    const uint16_t* __restrict__ B,    // K x N row-major bf16
    const float*    __restrict__ bias, // N
    void* __restrict__ Cout,
    int M, int N, int K) {
  __shared__ __align__(16) uint16_t As[2][BM * LSTR];
  __shared__ __align__(16) uint16_t Bs[2][BN * LSTR];  // transposed: [n][k]

  const int tid  = threadIdx.x;
  const int lane = tid & 31;
  const int wid  = tid >> 5;       // 0..7
  const int wm   = wid & 1;        // 2 wave-rows of 64
  const int wn   = wid >> 1;       // 4 wave-cols of 32
  const int half = lane >> 4;
  const int l16  = lane & 15;

  const int blockM = blockIdx.y * BM;
  const int blockN = blockIdx.x * BN;

  FragC acc[4][2];
  #pragma unroll
  for (int i = 0; i < 4; i++)
    #pragma unroll
    for (int j = 0; j < 2; j++)
      #pragma unroll
      for (int r = 0; r < 8; r++) acc[i][j].f[r] = 0.f;

  const int a_row = tid >> 1;            // 0..127
  const int a_col = (tid & 1) * 16;      // 0 or 16
  const int b_k   = tid >> 3;            // 0..31
  const int b_n   = (tid & 7) * 16;      // 0..112

  uint4 a0, a1;
  FragAB bu;
  auto fetch = [&](int kk) {
    const uint16_t* asrc = A + (size_t)(blockM + a_row) * K + kk + a_col;
    a0 = *reinterpret_cast<const uint4*>(asrc);
    a1 = *reinterpret_cast<const uint4*>(asrc + 8);
    const uint16_t* bsrc = B + (size_t)(kk + b_k) * N + blockN + b_n;
    bu.q[0] = *reinterpret_cast<const uint4*>(bsrc);
    bu.q[1] = *reinterpret_cast<const uint4*>(bsrc + 8);
  };
  auto stage = [&](int buf) {
    *reinterpret_cast<uint4*>(&As[buf][a_row * LSTR + a_col])     = a0;
    *reinterpret_cast<uint4*>(&As[buf][a_row * LSTR + a_col + 8]) = a1;
    #pragma unroll
    for (int i = 0; i < 16; i++)
      Bs[buf][(b_n + i) * LSTR + b_k] = bu.e[i];
  };

  fetch(0);
  stage(0);
  int cur = 0;

  for (int kk = 0; kk < K; kk += BK) {
    __syncthreads();                       // LDS[cur] ready; LDS[cur^1] free
    const bool more = (kk + BK) < K;
    if (more) fetch(kk + BK);              // global loads overlap WMMAs below

    FragAB afrag[4];
    #pragma unroll
    for (int mt = 0; mt < 4; mt++) {
      const uint16_t* p = &As[cur][(wm * 64 + mt * 16 + l16) * LSTR + half * 16];
      afrag[mt].q[0] = *reinterpret_cast<const uint4*>(p);
      afrag[mt].q[1] = *reinterpret_cast<const uint4*>(p + 8);
    }
    #pragma unroll
    for (int nt = 0; nt < 2; nt++) {
      FragAB bfrag;
      const uint16_t* p = &Bs[cur][(wn * 32 + nt * 16 + l16) * LSTR + half * 16];
      bfrag.q[0] = *reinterpret_cast<const uint4*>(p);
      bfrag.q[1] = *reinterpret_cast<const uint4*>(p + 8);
      #pragma unroll
      for (int mt = 0; mt < 4; mt++)
        acc[mt][nt].v = WMMA_BF16(afrag[mt].v, bfrag.v, acc[mt][nt].v);
    }

    if (more) stage(cur ^ 1);              // safe: other buffer, synced next iter
    cur ^= 1;
  }

  // ---- epilogue: bias + store ----
  #pragma unroll
  for (int mt = 0; mt < 4; mt++) {
    const int gm = blockM + wm * 64 + mt * 16 + half * 8;
    #pragma unroll
    for (int nt = 0; nt < 2; nt++) {
      const int gn = blockN + wn * 32 + nt * 16 + l16;
      const float bv = bias[gn];
      if (OUT_BF16) {
        uint16_t* C = (uint16_t*)Cout;
        #pragma unroll
        for (int r = 0; r < 8; r++)
          C[(size_t)(gm + r) * N + gn] = f2bf(acc[mt][nt].f[r] + bv);
      } else {
        float* C = (float*)Cout;
        #pragma unroll
        for (int r = 0; r < 8; r++)
          C[(size_t)(gm + r) * N + gn] = acc[mt][nt].f[r] + bv;
      }
    }
  }
}

// ---------------------------------------------------------------------------
// Flash attention: Out[i, outColBase..+1023] = softmax(Q K^T) V
// Q,K: [n][1024] bf16 row-major; Vt: [1024][n] bf16 (V transposed);
// Out: [n][2048] bf16 (concatenated buffer).
// One block = 16 query rows, 256 threads (8 waves). Key blocks of 128.
// Each wave: one 16x16 score tile, then a 128-wide slice of the PV product.
// Softmax is fully parallel: 16 lanes per row + xor-shuffle reduction.
// ---------------------------------------------------------------------------
#define FA_BC 128

__global__ __launch_bounds__(256) void k_flash(
    const uint16_t* __restrict__ Q,
    const uint16_t* __restrict__ Km,
    const uint16_t* __restrict__ Vt,
    uint16_t* __restrict__ Out,
    int n, int outColBase) {
  constexpr int Dh = 1024;
  constexpr int QS = Dh + 8;      // 1032 bf16 = 2064B stride (16B aligned)
  __shared__ __align__(16) uint16_t Qs[16 * QS];
  __shared__ __align__(16) float    Ss[16 * 132];
  __shared__ __align__(16) uint16_t Ps[16 * 136];
  __shared__ float m_run[16], l_run[16], alpha_s[16];

  const int tid  = threadIdx.x;
  const int lane = tid & 31;
  const int wid  = tid >> 5;
  const int half = lane >> 4;
  const int l16  = lane & 15;
  const int i0   = blockIdx.x * 16;

  // stage Q block (16 x 1024) into LDS: 64 bf16 per thread
  {
    const int row = tid >> 4;
    const int cb  = (tid & 15) * 64;
    const uint16_t* src = Q + (size_t)(i0 + row) * Dh + cb;
    uint16_t* dst = &Qs[row * QS + cb];
    #pragma unroll
    for (int i = 0; i < 8; i++)
      *reinterpret_cast<uint4*>(dst + i * 8) =
          *reinterpret_cast<const uint4*>(src + i * 8);
  }
  if (tid < 16) { m_run[tid] = -3.0e38f; l_run[tid] = 0.f; }
  __syncthreads();

  FragC oacc[8];
  #pragma unroll
  for (int nt = 0; nt < 8; nt++)
    #pragma unroll
    for (int r = 0; r < 8; r++) oacc[nt].f[r] = 0.f;

  for (int j0 = 0; j0 < n; j0 += FA_BC) {
    // prefetch next key block (K/V mostly L2-resident; cheap insurance)
    if (j0 + FA_BC < n) {
      __builtin_prefetch(Km + (size_t)(j0 + FA_BC + wid * 16 + l16) * Dh, 0, 3);
      __builtin_prefetch(Vt + (size_t)(wid * 128 + l16) * n + j0 + FA_BC, 0, 3);
    }

    // ---- S tile: this wave covers key cols j0 + wid*16 .. +15 ----
    FragC sacc;
    #pragma unroll
    for (int r = 0; r < 8; r++) sacc.f[r] = 0.f;
    const uint16_t* Kbase = Km + (size_t)(j0 + wid * 16 + l16) * Dh;
    for (int kk = 0; kk < Dh; kk += 32) {
      FragAB a, b;
      const uint16_t* ap = &Qs[l16 * QS + kk + half * 16];
      a.q[0] = *reinterpret_cast<const uint4*>(ap);
      a.q[1] = *reinterpret_cast<const uint4*>(ap + 8);
      const uint16_t* bp = Kbase + kk + half * 16;
      b.q[0] = *reinterpret_cast<const uint4*>(bp);
      b.q[1] = *reinterpret_cast<const uint4*>(bp + 8);
      sacc.v = WMMA_BF16(a.v, b.v, sacc.v);
    }
    #pragma unroll
    for (int r = 0; r < 8; r++)
      Ss[(half * 8 + r) * 132 + wid * 16 + l16] = sacc.f[r];
    __syncthreads();

    // ---- online softmax, fully parallel: 16 lanes per row, 8 cols per lane.
    // Row r = tid>>4 occupies one 16-lane half-group of a wave, so xor-shuffle
    // with masks 1,2,4,8 reduces within the row.
    {
      const int row = tid >> 4;
      const int sub = tid & 15;
      float sv[8];
      float vmax = -3.0e38f;
      const float* srow = &Ss[row * 132 + sub * 8];
      #pragma unroll
      for (int c = 0; c < 8; c++) { sv[c] = srow[c]; vmax = fmaxf(vmax, sv[c]); }
      #pragma unroll
      for (int s = 8; s >= 1; s >>= 1)
        vmax = fmaxf(vmax, __shfl_xor(vmax, s, 32));

      const float mold = m_run[row];
      const float mx = fmaxf(vmax, mold);
      Pack8 pk;
      float psum = 0.f;
      #pragma unroll
      for (int c = 0; c < 8; c++) {
        float p = __expf(sv[c] - mx);
        psum += p;
        pk.e[c] = f2bf(p);
      }
      *reinterpret_cast<uint4*>(&Ps[row * 136 + sub * 8]) = pk.q;
      #pragma unroll
      for (int s = 8; s >= 1; s >>= 1)
        psum += __shfl_xor(psum, s, 32);
      if (sub == 0) {
        const float al = __expf(mold - mx);
        alpha_s[row] = al;
        l_run[row] = l_run[row] * al + psum;
        m_run[row] = mx;
      }
    }
    __syncthreads();

    // ---- rescale accumulators, then P(16x128) @ V_j(128 x 128-slice) ----
    float al[8];
    #pragma unroll
    for (int r = 0; r < 8; r++) al[r] = alpha_s[half * 8 + r];
    #pragma unroll
    for (int nt = 0; nt < 8; nt++)
      #pragma unroll
      for (int r = 0; r < 8; r++) oacc[nt].f[r] *= al[r];

    const int c0 = wid * 128;   // this wave's 128-wide output slice
    #pragma unroll
    for (int ks = 0; ks < 4; ks++) {
      FragAB a;
      const uint16_t* ap = &Ps[l16 * 136 + ks * 32 + half * 16];
      a.q[0] = *reinterpret_cast<const uint4*>(ap);
      a.q[1] = *reinterpret_cast<const uint4*>(ap + 8);
      #pragma unroll
      for (int nt = 0; nt < 8; nt++) {
        FragAB b;
        const uint16_t* bp =
            Vt + (size_t)(c0 + nt * 16 + l16) * n + j0 + ks * 32 + half * 16;
        b.q[0] = *reinterpret_cast<const uint4*>(bp);
        b.q[1] = *reinterpret_cast<const uint4*>(bp + 8);
        oacc[nt].v = WMMA_BF16(a.v, b.v, oacc[nt].v);
      }
    }
    __syncthreads();
  }

  // ---- normalize and store bf16 into fused buffer ----
  float inv[8];
  #pragma unroll
  for (int r = 0; r < 8; r++) inv[r] = 1.f / l_run[half * 8 + r];
  #pragma unroll
  for (int nt = 0; nt < 8; nt++) {
    const int gn = outColBase + wid * 128 + nt * 16 + l16;
    #pragma unroll
    for (int r = 0; r < 8; r++)
      Out[(size_t)(i0 + half * 8 + r) * 2048 + gn] = f2bf(oacc[nt].f[r] * inv[r]);
  }
}

// ---------------------------------------------------------------------------
// Host orchestration
// ---------------------------------------------------------------------------
extern "C" void kernel_launch(void* const* d_in, const int* in_sizes, int n_in,
                              void* d_out, int out_size, void* d_ws, size_t ws_size,
                              hipStream_t stream) {
  (void)in_sizes; (void)n_in; (void)out_size; (void)ws_size;
  constexpr int N = 8192, D = 1024;

  const float* img = (const float*)d_in[0];
  const float* tab = (const float*)d_in[1];
  const float* Wqi = (const float*)d_in[2];  const float* bqi = (const float*)d_in[3];
  const float* Wkt = (const float*)d_in[4];  const float* bkt = (const float*)d_in[5];
  const float* Wvt = (const float*)d_in[6];  const float* bvt = (const float*)d_in[7];
  const float* Wqt = (const float*)d_in[8];  const float* bqt = (const float*)d_in[9];
  const float* Wki = (const float*)d_in[10]; const float* bki = (const float*)d_in[11];
  const float* Wvi = (const float*)d_in[12]; const float* bvi = (const float*)d_in[13];
  const float* Wo  = (const float*)d_in[14]; const float* bo  = (const float*)d_in[15];

  char* ws = (char*)d_ws;
  size_t off = 0;
  auto alloc = [&](size_t bytes) -> uint16_t* {
    uint16_t* p = (uint16_t*)(ws + off);
    off += (bytes + 255) & ~(size_t)255;
    return p;
  };
  const size_t ND = (size_t)N * D * 2, DD = (size_t)D * D * 2;
  uint16_t* bImg = alloc(ND);
  uint16_t* bTab = alloc(ND);
  uint16_t* wqi = alloc(DD); uint16_t* wkt = alloc(DD); uint16_t* wvt = alloc(DD);
  uint16_t* wqt = alloc(DD); uint16_t* wki = alloc(DD); uint16_t* wvi = alloc(DD);
  uint16_t* wo  = alloc(4 * DD);
  uint16_t* Qi  = alloc(ND); uint16_t* Kt = alloc(ND); uint16_t* Vta = alloc(ND);
  uint16_t* Qt  = alloc(ND); uint16_t* Ki = alloc(ND); uint16_t* Vim = alloc(ND);
  uint16_t* VtaT = alloc(ND); uint16_t* VimT = alloc(ND);
  uint16_t* Fused = alloc((size_t)N * 2 * D * 2);

  auto conv = [&](const float* s, uint16_t* d, int n) {
    k_f32_to_bf16<<<(n + 255) / 256, 256, 0, stream>>>(s, d, n);
  };
  conv(img, bImg, N * D);  conv(tab, bTab, N * D);
  conv(Wqi, wqi, D * D);   conv(Wkt, wkt, D * D);  conv(Wvt, wvt, D * D);
  conv(Wqt, wqt, D * D);   conv(Wki, wki, D * D);  conv(Wvi, wvi, D * D);
  conv(Wo,  wo,  4 * D * D);

  // projections: [N,D] @ [D,D] + b -> bf16
  dim3 gP(D / BN, N / BM);
  k_gemm_bf16<1><<<gP, 256, 0, stream>>>(bImg, wqi, bqi, Qi,  N, D, D);
  k_gemm_bf16<1><<<gP, 256, 0, stream>>>(bTab, wkt, bkt, Kt,  N, D, D);
  k_gemm_bf16<1><<<gP, 256, 0, stream>>>(bTab, wvt, bvt, Vta, N, D, D);
  k_gemm_bf16<1><<<gP, 256, 0, stream>>>(bTab, wqt, bqt, Qt,  N, D, D);
  k_gemm_bf16<1><<<gP, 256, 0, stream>>>(bImg, wki, bki, Ki,  N, D, D);
  k_gemm_bf16<1><<<gP, 256, 0, stream>>>(bImg, wvi, bvi, Vim, N, D, D);

  // transpose V's: [N,D] -> [D,N]
  dim3 gT(D / 32, N / 32);
  k_transpose_bf16<<<gT, 256, 0, stream>>>(Vta, VtaT, N, D);
  k_transpose_bf16<<<gT, 256, 0, stream>>>(Vim, VimT, N, D);

  // branch 1: softmax(Qi Kt^T) Vta -> attended_tabular -> cols [D, 2D)
  k_flash<<<N / 16, 256, 0, stream>>>(Qi, Kt, VtaT, Fused, N, D);
  // branch 2: softmax(Qt Ki^T) Vim -> attended_image -> cols [0, D)
  k_flash<<<N / 16, 256, 0, stream>>>(Qt, Ki, VimT, Fused, N, 0);

  // output: [N,2D] @ [2D,2D] + bo -> f32 d_out
  dim3 gO(2 * D / BN, N / BM);
  k_gemm_bf16<0><<<gO, 256, 0, stream>>>(Fused, wo, bo, d_out, N, 2 * D, 2 * D);
}
